// MultiHeadAttention_29446295781411
// MI455X (gfx1250) — matrix operand order repaired and checked by reading, not compiled, not run on hardware
//
#include <hip/hip_runtime.h>
#include <hip/hip_bf16.h>

typedef __bf16 bf16;
typedef __attribute__((ext_vector_type(16))) __bf16 v16bf;
typedef __attribute__((ext_vector_type(8)))  __bf16 v8bf;
typedef __attribute__((ext_vector_type(4)))  __bf16 v4bf;
typedef __attribute__((ext_vector_type(8)))  float  v8f;

#define D_MODEL  1024
#define NUM_HEAD 16
#define HEAD_DIM 64
#define BATCH    4
#define SEQ      2048
#define M_TOT    (BATCH * SEQ)   // 8192 tokens

// ---------------------------------------------------------------------------
// Fragment loaders.  WMMA bf16 16x16x32 (CDNA5 ISA 7.12.2):
//  A (16x32, MxK): lane l -> row M = l%16; half h = l/16;
//     VGPR i: K = (i/4)*16 + h*8 + (i%4)*2 (+0,+1)
//     => per lane two contiguous 16B runs at K offsets  h*8  and  16+h*8.
//  B (32x16, KxN): lane l -> col N = l%16; half h = l/16;
//     VGPR i: K = (i/4)*16 + h*8 + (i%4)*2 (+0,+1)
//     => per lane two contiguous 16B runs at K offsets  h*8  and  16+h*8.
// ---------------------------------------------------------------------------
static __device__ inline v8bf ld16B(const bf16* p) {
    return *reinterpret_cast<const v8bf*>(p);
}

static __device__ inline v16bf frag_cat(v8bf lo, v8bf hi) {
    return __builtin_shufflevector(lo, hi, 0,1,2,3,4,5,6,7,8,9,10,11,12,13,14,15);
}

// A fragment: rows m0..m0+15, K-dims k0..k0+31 from row-major [.,ld]
static __device__ inline v16bf load_a(const bf16* __restrict__ base, int ld,
                                      int m0, int k0, int ml, int hl) {
    const bf16* r = base + (long)(m0 + ml) * ld + k0 + hl * 8;
    return frag_cat(ld16B(r), ld16B(r + 16));
}

// B fragment for B[k,n] = W[n,k], W row-major [N][ld]: cols n0..n0+15, K k0..k0+31
static __device__ inline v16bf load_b(const bf16* __restrict__ base, int ld,
                                      int n0, int k0, int ml, int hl) {
    const bf16* r = base + (long)(n0 + ml) * ld + k0 + hl * 8;
    return frag_cat(ld16B(r), ld16B(r + 16));
}

static __device__ inline v8f wmma_bf16(v16bf a, v16bf b, v8f c) {
    return __builtin_amdgcn_wmma_f32_16x16x32_bf16(false, a, false, b,
                                                   (short)0, c, false, false);
}

// ---------------------------------------------------------------------------
// fp32 -> bf16 conversion (vectorized x4)
// ---------------------------------------------------------------------------
__global__ void cvt_bf16_kernel(const float* __restrict__ src,
                                bf16* __restrict__ dst, int n) {
    int i = (blockIdx.x * blockDim.x + threadIdx.x) * 4;
    if (i < n) {
        float4 v = *reinterpret_cast<const float4*>(src + i);
        v4bf o;
        o[0] = (bf16)v.x; o[1] = (bf16)v.y; o[2] = (bf16)v.z; o[3] = (bf16)v.w;
        *reinterpret_cast<v4bf*>(dst + i) = o;
    }
}

// ---------------------------------------------------------------------------
// GEMM: C = scale * (A[MxK] * W[NxK]^T)   (nn.Linear: out = x @ W.T)
// Block = 256 threads = 8 waves (4 along M x 2 along N).
// Wave tile = 32x64 (2x4 WMMA tiles) -> block tile 128x128.
// mode 0: store bf16 row-major [M][N]
// mode 1: store bf16 transposed per head: Vt[b][h][d][t]  (contiguous b128)
// mode 2: store fp32 row-major [M][N]
// ---------------------------------------------------------------------------
__global__ __launch_bounds__(256) void gemm_bf16_kernel(
    const bf16* __restrict__ A, const bf16* __restrict__ W,
    void* __restrict__ C, int M, int N, int K, float scale, int mode) {

    const int tid  = threadIdx.x;
    const int lane = tid & 31;
    const int wave = tid >> 5;
    const int ml   = lane & 15;
    const int hl   = lane >> 4;
    const int wm   = wave & 3;           // 4 waves along M
    const int wn   = wave >> 2;          // 2 waves along N
    const int m0   = blockIdx.y * 128 + wm * 32;
    const int n0   = blockIdx.x * 128 + wn * 64;

    v8f acc[2][4] = {};

    for (int k0 = 0; k0 < K; k0 += 32) {
        v16bf a0 = load_a(A, K, m0,      k0, ml, hl);
        v16bf a1 = load_a(A, K, m0 + 16, k0, ml, hl);
        v16bf b0 = load_b(W, K, n0,      k0, ml, hl);
        v16bf b1 = load_b(W, K, n0 + 16, k0, ml, hl);
        v16bf b2 = load_b(W, K, n0 + 32, k0, ml, hl);
        v16bf b3 = load_b(W, K, n0 + 48, k0, ml, hl);
        acc[0][0] = wmma_bf16(a0, b0, acc[0][0]);
        acc[0][1] = wmma_bf16(a0, b1, acc[0][1]);
        acc[0][2] = wmma_bf16(a0, b2, acc[0][2]);
        acc[0][3] = wmma_bf16(a0, b3, acc[0][3]);
        acc[1][0] = wmma_bf16(a1, b0, acc[1][0]);
        acc[1][1] = wmma_bf16(a1, b1, acc[1][1]);
        acc[1][2] = wmma_bf16(a1, b2, acc[1][2]);
        acc[1][3] = wmma_bf16(a1, b3, acc[1][3]);
    }

    #pragma unroll
    for (int ti = 0; ti < 2; ++ti) {
        #pragma unroll
        for (int tj = 0; tj < 4; ++tj) {
            const int row0 = m0 + ti * 16 + hl * 8;     // rows row0..row0+7
            const int col  = n0 + tj * 16 + ml;         // fixed column per lane
            if (mode == 0) {
                bf16* Cb = (bf16*)C;
                #pragma unroll
                for (int i = 0; i < 8; ++i)
                    Cb[(long)(row0 + i) * N + col] = (bf16)(acc[ti][tj][i] * scale);
            } else if (mode == 2) {
                float* Cf = (float*)C;
                #pragma unroll
                for (int i = 0; i < 8; ++i)
                    Cf[(long)(row0 + i) * N + col] = acc[ti][tj][i] * scale;
            } else {
                // Vt[b][h][d][t] : C-frag rows are consecutive tokens -> b128 store
                const int b  = row0 / SEQ;
                const int t0 = row0 % SEQ;
                const int hd = col / HEAD_DIM;
                const int d  = col % HEAD_DIM;
                v8bf pk;
                #pragma unroll
                for (int i = 0; i < 8; ++i) pk[i] = (bf16)(acc[ti][tj][i] * scale);
                bf16* Vt = (bf16*)C;
                *reinterpret_cast<v8bf*>(
                    Vt + ((long)(b * NUM_HEAD + hd) * HEAD_DIM + d) * SEQ + t0) = pk;
            }
        }
    }
}

// ---------------------------------------------------------------------------
// Flash attention (causal), one wave per 32-query tile (two 16-row WMMA tiles
// sharing every K/V fragment -> 2x operand reuse vs 16-row waves).
// Q: [B*S][D_MODEL] bf16 (pre-scaled by 1/sqrt(D_MODEL))
// K: [B*S][D_MODEL] bf16
// Vt: [B][H][HEAD_DIM][S] bf16 (transposed so V B-fragments are contiguous)
// O: [B*S][D_MODEL] bf16 (attn output, pre out-projection)
// ---------------------------------------------------------------------------
__global__ __launch_bounds__(256) void attn_flash_kernel(
    const bf16* __restrict__ Q, const bf16* __restrict__ Km,
    const bf16* __restrict__ Vt, bf16* __restrict__ O) {

    // per-wave P staging: two 16x32 bf16 tiles (C-layout -> A-layout)
    __shared__ __bf16 Pb[8][2 * 16 * 32];

    const int tid  = threadIdx.x;
    const int lane = tid & 31;
    const int wave = tid >> 5;
    const int ml   = lane & 15;
    const int hl   = lane >> 4;
    const int bh   = blockIdx.y;
    const int b    = bh / NUM_HEAD;
    const int hd   = bh % NUM_HEAD;
    const int q0   = blockIdx.x * 256 + wave * 32;   // 32 queries per wave

    const bf16* Qh = Q  + (long)b * SEQ * D_MODEL + hd * HEAD_DIM;
    const bf16* Kh = Km + (long)b * SEQ * D_MODEL + hd * HEAD_DIM;
    const bf16* Vh = Vt + (long)(b * NUM_HEAD + hd) * HEAD_DIM * SEQ;

    // Q fragments reused across the whole key loop
    const v16bf qA0 = load_a(Qh, D_MODEL, q0,      0,  ml, hl);
    const v16bf qA1 = load_a(Qh, D_MODEL, q0,      32, ml, hl);
    const v16bf qB0 = load_a(Qh, D_MODEL, q0 + 16, 0,  ml, hl);
    const v16bf qB1 = load_a(Qh, D_MODEL, q0 + 16, 32, ml, hl);

    v8f oA[4] = {}, oB[4] = {};
    float mA[8], lA[8], mB[8], lB[8];
    #pragma unroll
    for (int i = 0; i < 8; ++i) {
        mA[i] = -1e30f; lA[i] = 0.0f;
        mB[i] = -1e30f; lB[i] = 0.0f;
    }

    const int kend = q0 + 31;                   // causal: keys <= query
    for (int kb = 0; kb <= kend; kb += 32) {
        // score tiles: S = Q (2x 16x64) . K^T (64x32)
        v8f sA0 = {}, sA1 = {}, sB0 = {}, sB1 = {};
        {
            v16bf kb0 = load_b(Kh, D_MODEL, kb,      0,  ml, hl);
            v16bf kb1 = load_b(Kh, D_MODEL, kb,      32, ml, hl);
            v16bf kc0 = load_b(Kh, D_MODEL, kb + 16, 0,  ml, hl);
            v16bf kc1 = load_b(Kh, D_MODEL, kb + 16, 32, ml, hl);
            sA0 = wmma_bf16(qA0, kb0, sA0);
            sA0 = wmma_bf16(qA1, kb1, sA0);
            sA1 = wmma_bf16(qA0, kc0, sA1);
            sA1 = wmma_bf16(qA1, kc1, sA1);
            sB0 = wmma_bf16(qB0, kb0, sB0);
            sB0 = wmma_bf16(qB1, kb1, sB0);
            sB1 = wmma_bf16(qB0, kc0, sB1);
            sB1 = wmma_bf16(qB1, kc1, sB1);
        }

        // causal mask + online softmax. C-frag: element i <-> row (+hl*8+i),
        // lane%16 <-> column; row reductions = 16-lane xor butterflies.
        const int key0 = kb + ml;
        const int key1 = kb + 16 + ml;

        #pragma unroll
        for (int t = 0; t < 2; ++t) {           // t=0: tile A, t=1: tile B
            v8f&   s0   = t ? sB0 : sA0;
            v8f&   s1   = t ? sB1 : sA1;
            float* mrow = t ? mB : mA;
            float* lrow = t ? lB : lA;
            v8f*   o    = t ? oB : oA;
            const int qb = q0 + t * 16;
            float alpha[8], p0[8], p1[8];
            #pragma unroll
            for (int i = 0; i < 8; ++i) {
                const int qrow = qb + hl * 8 + i;
                float v0 = (key0 > qrow) ? -1e9f : s0[i];
                float v1 = (key1 > qrow) ? -1e9f : s1[i];
                float mx = fmaxf(v0, v1);
                mx = fmaxf(mx, __shfl_xor(mx, 1));
                mx = fmaxf(mx, __shfl_xor(mx, 2));
                mx = fmaxf(mx, __shfl_xor(mx, 4));
                mx = fmaxf(mx, __shfl_xor(mx, 8));
                const float mn = fmaxf(mrow[i], mx);
                const float al = __expf(mrow[i] - mn);
                const float e0 = __expf(v0 - mn);
                const float e1 = __expf(v1 - mn);
                float rs = e0 + e1;
                rs += __shfl_xor(rs, 1);
                rs += __shfl_xor(rs, 2);
                rs += __shfl_xor(rs, 4);
                rs += __shfl_xor(rs, 8);
                lrow[i] = lrow[i] * al + rs;
                mrow[i] = mn;
                alpha[i] = al; p0[i] = e0; p1[i] = e1;
            }
            // rescale running output and stage P tile to LDS
            #pragma unroll
            for (int dt = 0; dt < 4; ++dt)
                #pragma unroll
                for (int i = 0; i < 8; ++i) o[dt][i] *= alpha[i];
            #pragma unroll
            for (int i = 0; i < 8; ++i) {
                Pb[wave][t * 512 + (hl * 8 + i) * 32 + ml]      = (bf16)p0[i];
                Pb[wave][t * 512 + (hl * 8 + i) * 32 + 16 + ml] = (bf16)p1[i];
            }
        }

        asm volatile("s_wait_dscnt 0" ::: "memory");
        const __bf16* prA = &Pb[wave][ml * 32 + hl * 8];
        v16bf paA = frag_cat(*reinterpret_cast<const v8bf*>(prA),
                             *reinterpret_cast<const v8bf*>(prA + 16));
        const __bf16* prB = prA + 512;
        v16bf paB = frag_cat(*reinterpret_cast<const v8bf*>(prB),
                             *reinterpret_cast<const v8bf*>(prB + 16));

        // O += P (16x32) . V (32x64): each V B-frag feeds both query tiles
        #pragma unroll
        for (int dt = 0; dt < 4; ++dt) {
            v16bf vb = load_b(Vh, SEQ, dt * 16, kb, ml, hl);
            oA[dt] = wmma_bf16(paA, vb, oA[dt]);
            oB[dt] = wmma_bf16(paB, vb, oB[dt]);
        }
    }

    // normalize and write attn output (row-major [token][h*64+d], bf16)
    #pragma unroll
    for (int i = 0; i < 8; ++i) { lA[i] = 1.0f / lA[i]; lB[i] = 1.0f / lB[i]; }
    bf16* Oh = O + (long)b * SEQ * D_MODEL + hd * HEAD_DIM;
    #pragma unroll
    for (int dt = 0; dt < 4; ++dt)
        #pragma unroll
        for (int i = 0; i < 8; ++i) {
            Oh[(long)(q0 + hl * 8 + i) * D_MODEL + dt * 16 + ml] =
                (bf16)(oA[dt][i] * lA[i]);
            Oh[(long)(q0 + 16 + hl * 8 + i) * D_MODEL + dt * 16 + ml] =
                (bf16)(oB[dt][i] * lB[i]);
        }
}

// ---------------------------------------------------------------------------
// Host launcher
// ---------------------------------------------------------------------------
extern "C" void kernel_launch(void* const* d_in, const int* in_sizes, int n_in,
                              void* d_out, int out_size, void* d_ws, size_t ws_size,
                              hipStream_t stream) {
    const float* x  = (const float*)d_in[0];
    // d_in[1] is the boolean causal mask; mask is computed analytically in-kernel.
    const float* wq = (const float*)d_in[2];
    const float* wk = (const float*)d_in[3];
    const float* wv = (const float*)d_in[4];
    const float* wo = (const float*)d_in[5];

    constexpr size_t SZ_X = (size_t)M_TOT * D_MODEL * sizeof(bf16);   // 16 MB
    constexpr size_t SZ_W = (size_t)D_MODEL * D_MODEL * sizeof(bf16); //  2 MB

    char* ws = (char*)d_ws;
    bf16* Xb  = (bf16*)(ws);
    bf16* Wqb = (bf16*)(ws + SZ_X);
    bf16* Wkb = (bf16*)(ws + SZ_X + SZ_W);
    bf16* Wvb = (bf16*)(ws + SZ_X + 2 * SZ_W);
    bf16* Wob = (bf16*)(ws + SZ_X + 3 * SZ_W);
    bf16* Qb  = (bf16*)(ws + SZ_X + 4 * SZ_W);
    bf16* Kb  = (bf16*)(ws + 2 * SZ_X + 4 * SZ_W);
    bf16* Vtb = (bf16*)(ws + 3 * SZ_X + 4 * SZ_W);
    bf16* Ab  = (bf16*)(ws + 4 * SZ_X + 4 * SZ_W);

    // fp32 -> bf16 conversions
    {
        const int nx = M_TOT * D_MODEL;
        cvt_bf16_kernel<<<(nx / 4 + 255) / 256, 256, 0, stream>>>(x, Xb, nx);
        const int nw = D_MODEL * D_MODEL;
        const int gb = (nw / 4 + 255) / 256;
        cvt_bf16_kernel<<<gb, 256, 0, stream>>>(wq, Wqb, nw);
        cvt_bf16_kernel<<<gb, 256, 0, stream>>>(wk, Wkb, nw);
        cvt_bf16_kernel<<<gb, 256, 0, stream>>>(wv, Wvb, nw);
        cvt_bf16_kernel<<<gb, 256, 0, stream>>>(wo, Wob, nw);
    }

    const dim3 ggrid(D_MODEL / 128, M_TOT / 128);  // (8, 64)
    const float qscale = 0.03125f;                 // 1/sqrt(1024)

    // QKV projections (Q pre-scaled; V stored transposed per head)
    gemm_bf16_kernel<<<ggrid, 256, 0, stream>>>(Xb, Wqb, Qb,
        M_TOT, D_MODEL, D_MODEL, qscale, 0);
    gemm_bf16_kernel<<<ggrid, 256, 0, stream>>>(Xb, Wkb, Kb,
        M_TOT, D_MODEL, D_MODEL, 1.0f, 0);
    gemm_bf16_kernel<<<ggrid, 256, 0, stream>>>(Xb, Wvb, Vtb,
        M_TOT, D_MODEL, D_MODEL, 1.0f, 1);

    // causal flash attention (32 queries per wave, 256 per block)
    attn_flash_kernel<<<dim3(SEQ / 256, BATCH * NUM_HEAD), 256, 0, stream>>>(
        Qb, Kb, Vtb, Ab);

    // output projection -> fp32 d_out
    gemm_bf16_kernel<<<ggrid, 256, 0, stream>>>(Ab, Wob, (float*)d_out,
        M_TOT, D_MODEL, D_MODEL, 1.0f, 2);
}
